// ANIMAOneExact_86887188398422
// MI455X (gfx1250) — matrix operand order repaired and checked by reading, not compiled, not run on hardware
//
#include <hip/hip_runtime.h>
#include <cstdint>
#include <cstddef>

// ============================================================================
// ANIMA recurrent scan on MI455X (gfx1250): wave32 + v_wmma_f32_16x16x32_f16.
// One wave owns a 16-row batch tile and scans all SEQ steps; recurrent state
// (M, Dst) is carried in per-wave LDS staging buffers (f16). Weights are
// staged once per workgroup into LDS as f16, zero-padded in K to multiples of
// 32 (and N of W_out to 16) so A-side K padding never needs clearing.
// Activations use native v_rcp_f32/v_exp_f32 (TRANS32 co-executes with WMMA).
// ============================================================================

typedef __attribute__((ext_vector_type(16))) _Float16 v16h;
typedef __attribute__((ext_vector_type(8)))  float    v8f;

#define BATCH 2048
#define SEQ   1024
#define SDIM  8
#define ODIM  4

#define WAVES 4
#define TPB   (WAVES * 32)
#define ROWS_PER_WAVE 16
#define ROWS_PER_WG   (WAVES * ROWS_PER_WAVE)
#define NBLOCKS (BATCH / ROWS_PER_WG)

// ---- LDS layout (offsets/sizes in _Float16 elements) ----
#define SZ_WS (48 * 32)     // W_sense   48 x (8->32)
#define SZ_WC (32 * 64)     // W_compress 32 x (48->64)
#define SZ_WE (48 * 32)     // W_expand  48 x 32
#define SZ_WG (48 * 96)     // W_z / W_r / W_h  48 x 96
#define SZ_WB (48 * 160)    // W_int / W_phi    48 x (144->160)
#define SZ_WO (16 * 64)     // W_out  (4->16) x (48->64)

#define OFF_WS 0
#define OFF_WC (OFF_WS + SZ_WS)
#define OFF_WE (OFF_WC + SZ_WC)
#define OFF_WZ (OFF_WE + SZ_WE)
#define OFF_WR (OFF_WZ + SZ_WG)
#define OFF_WH (OFF_WR + SZ_WG)
#define OFF_WI (OFF_WH + SZ_WG)
#define OFF_WP (OFF_WI + SZ_WB)
#define OFF_WO (OFF_WP + SZ_WB)
#define OFF_BIAS (OFF_WO + SZ_WO)       // float (f32) region starts here
#define NBIAS 384                        // padded biases, in floats
#define OFF_ACT (OFF_BIAS + 2 * NBIAS)   // per-wave activation staging
#define ACT_PER_WAVE 6144                // 12 KB per wave
#define A_BUF32 0                        // 16x32: x tile / compressed / action scratch
#define A_COMB  512                      // 16x96: [sensed | M]
#define A_COMBH 2048                     // 16x96: [sensed | r*M]
#define A_ALLST 3584                     // 16x160: [S_new | M_new | Dst | pad]
#define LDS_BYTES ((OFF_ACT + WAVES * ACT_PER_WAVE) * 2)   // 121344 B < 320 KB

// bias sub-offsets (in floats)
#define B_S 0
#define B_C 48
#define B_E 80
#define B_Z 128
#define B_R 176
#define B_H 224
#define B_I 272
#define B_P 320
#define B_O 368

#define LDS_FENCE() asm volatile("s_wait_dscnt 0" ::: "memory")

// ---------------------------------------------------------------------------
// A fragment (16x32 f16, M per lane): lane half 0 -> K = k0..k0+15;
// lane half 1 -> K = k0+16..31.  p = base + m*stride + k0 + half*8.
__device__ __forceinline__ v16h frag_a(const _Float16* p) {
  union { uint4 u[2]; v16h h; } t;
  p += ((threadIdx.x & 31) >> 4) * 8;
  t.u[0] = *(const uint4*)(p);
  t.u[1] = *(const uint4*)(p + 8);
  return t.h;
}
// B fragment (32x16 f16, N per lane): lanes 0-15 hold K=0..15, lanes 16-31
// K=16..31 (contiguous along K of W^T = row of W).  p = W + n*stride + k0 + half*16.
__device__ __forceinline__ v16h frag_b(const _Float16* p) {
  union { uint4 u[2]; v16h h; } t;
  t.u[0] = *(const uint4*)(p);
  t.u[1] = *(const uint4*)(p + 8);
  return t.h;
}

__device__ __forceinline__ v8f wmma16(v16h a, v16h b, v8f c) {
  return __builtin_amdgcn_wmma_f32_16x16x32_f16(false, a, false, b,
                                                (short)0, c, false, false);
}

__device__ __forceinline__ v8f splat8(float x) {
  v8f v;
#pragma unroll
  for (int i = 0; i < 8; ++i) v[i] = x;
  return v;
}

// Native-rate activations: v_exp_f32 + v_rcp_f32 (TRANS32, co-execute w/ WMMA).
__device__ __forceinline__ float sigm_f(float x) {
  return __builtin_amdgcn_rcpf(1.0f + __expf(-x));
}
__device__ __forceinline__ float tanh_f(float x) {
  // tanh(x) = 1 - 2/(1 + e^{2x})
  return 1.0f - 2.0f * __builtin_amdgcn_rcpf(1.0f + __expf(2.0f * x));
}

__device__ __forceinline__ void vtanh(v8f& v) {
#pragma unroll
  for (int i = 0; i < 8; ++i) v[i] = tanh_f(v[i]);
}
__device__ __forceinline__ void vsigm(v8f& v) {
#pragma unroll
  for (int i = 0; i < 8; ++i) v[i] = sigm_f(v[i]);
}

// C/D fragment layout (§7.12.2): VGPR i -> row half*8+i, lane -> column n.
__device__ __forceinline__ void cstore(_Float16* buf, int stride, int col, int lh, v8f v) {
#pragma unroll
  for (int i = 0; i < 8; ++i) buf[(lh * 8 + i) * stride + col] = (_Float16)v[i];
}
__device__ __forceinline__ v8f cload(const _Float16* buf, int stride, int col, int lh) {
  v8f v;
#pragma unroll
  for (int i = 0; i < 8; ++i) v[i] = (float)buf[(lh * 8 + i) * stride + col];
  return v;
}

// f32 -> f16 weight staging with zero padding (src is W[n][k], row-major).
__device__ inline void stage_w(_Float16* dst, const float* src,
                               int N, int K, int Np, int Kp, int tid) {
  for (int idx = tid; idx < Np * Kp; idx += TPB) {
    int n = idx / Kp, k = idx - n * Kp;
    float v = (n < N && k < K) ? src[n * K + k] : 0.0f;
    dst[idx] = (_Float16)v;
  }
}
__device__ inline void stage_b(float* dst, const float* src, int n, int np, int tid) {
  for (int idx = tid; idx < np; idx += TPB) dst[idx] = (idx < n) ? src[idx] : 0.0f;
}

__global__ void __launch_bounds__(TPB, 1)
anima_scan_kernel(const float* __restrict__ x,
                  const float* __restrict__ Wse, const float* __restrict__ bse,
                  const float* __restrict__ Wco, const float* __restrict__ bco,
                  const float* __restrict__ Wex, const float* __restrict__ bex,
                  const float* __restrict__ Wz,  const float* __restrict__ bz,
                  const float* __restrict__ Wr,  const float* __restrict__ br,
                  const float* __restrict__ Wh,  const float* __restrict__ bh,
                  const float* __restrict__ Wi,  const float* __restrict__ bi,
                  const float* __restrict__ Wp,  const float* __restrict__ bp,
                  const float* __restrict__ Wo,  const float* __restrict__ bo,
                  float* __restrict__ out) {
  extern __shared__ _Float16 smem[];
  const int tid  = threadIdx.x;
  const int lane = tid & 31;
  const int wid  = tid >> 5;
  const int nidx = lane & 15;   // N index (B/C frags) and M row (A frags)
  const int lh   = lane >> 4;   // lane half

  // ---- one-time weight / bias staging into LDS ----
  stage_w(smem + OFF_WS, Wse, 48, 8,   48, 32,  tid);
  stage_w(smem + OFF_WC, Wco, 32, 48,  32, 64,  tid);
  stage_w(smem + OFF_WE, Wex, 48, 32,  48, 32,  tid);
  stage_w(smem + OFF_WZ, Wz,  48, 96,  48, 96,  tid);
  stage_w(smem + OFF_WR, Wr,  48, 96,  48, 96,  tid);
  stage_w(smem + OFF_WH, Wh,  48, 96,  48, 96,  tid);
  stage_w(smem + OFF_WI, Wi,  48, 144, 48, 160, tid);
  stage_w(smem + OFF_WP, Wp,  48, 144, 48, 160, tid);
  stage_w(smem + OFF_WO, Wo,  4,  48,  16, 64,  tid);
  float* bias = (float*)(smem + OFF_BIAS);
  stage_b(bias + B_S, bse, 48, 48, tid);
  stage_b(bias + B_C, bco, 32, 32, tid);
  stage_b(bias + B_E, bex, 48, 48, tid);
  stage_b(bias + B_Z, bz,  48, 48, tid);
  stage_b(bias + B_R, br,  48, 48, tid);
  stage_b(bias + B_H, bh,  48, 48, tid);
  stage_b(bias + B_I, bi,  48, 48, tid);
  stage_b(bias + B_P, bp,  48, 48, tid);
  stage_b(bias + B_O, bo,  4,  16, tid);

  _Float16* act   = smem + OFF_ACT + wid * ACT_PER_WAVE;
  _Float16* buf32 = act + A_BUF32;
  _Float16* comb  = act + A_COMB;
  _Float16* combH = act + A_COMBH;
  _Float16* allst = act + A_ALLST;
  for (int i = lane; i < ACT_PER_WAVE; i += 32) act[i] = (_Float16)0.0f;  // M=0, D=0

  __syncthreads();

  const _Float16* pWs = smem + OFF_WS;
  const _Float16* pWc = smem + OFF_WC;
  const _Float16* pWe = smem + OFF_WE;
  const _Float16* pWz = smem + OFF_WZ;
  const _Float16* pWr = smem + OFF_WR;
  const _Float16* pWh = smem + OFF_WH;
  const _Float16* pWi = smem + OFF_WI;
  const _Float16* pWp = smem + OFF_WP;
  const _Float16* pWo = smem + OFF_WO;

  const int row0 = blockIdx.x * ROWS_PER_WG + wid * ROWS_PER_WAVE;

#pragma unroll 1
  for (int t = 0; t < SEQ; ++t) {
    // (a) sensory tile -> buf32[:,0:8]  (cols 8..31 are garbage * zero weights)
    {
      const int r = lane >> 1, c = (lane & 1) * 4;
      const float4 xv = *(const float4*)(x + (((size_t)(row0 + r)) * SEQ + (size_t)t) * SDIM + c);
      _Float16* d = buf32 + r * 32 + c;
      d[0] = (_Float16)xv.x; d[1] = (_Float16)xv.y;
      d[2] = (_Float16)xv.z; d[3] = (_Float16)xv.w;
    }
    LDS_FENCE();

    // (b) sensed = tanh(x @ Wse^T + b) -> comb[:,0:48], combH[:,0:48]
    {
      const v16h ax = frag_a(buf32 + nidx * 32 + lh * 8);
#pragma unroll
      for (int nt = 0; nt < 3; ++nt) {
        v8f c = splat8(bias[B_S + nt * 16 + nidx]);
        c = wmma16(ax, frag_b(pWs + (nt * 16 + nidx) * 32 + lh * 16), c);
        vtanh(c);
        cstore(comb,  96, nt * 16 + nidx, lh, c);
        cstore(combH, 96, nt * 16 + nidx, lh, c);
      }
    }
    LDS_FENCE();

    // (c) compressed = tanh(sensed @ Wco^T + b) -> buf32[:,0:32]
    {
      const v16h a0 = frag_a(comb + nidx * 96 + 0  + lh * 8);
      const v16h a1 = frag_a(comb + nidx * 96 + 32 + lh * 8);  // K 32..63: M region * zero wts
#pragma unroll
      for (int nt = 0; nt < 2; ++nt) {
        v8f c = splat8(bias[B_C + nt * 16 + nidx]);
        c = wmma16(a0, frag_b(pWc + (nt * 16 + nidx) * 64 + 0  + lh * 16), c);
        c = wmma16(a1, frag_b(pWc + (nt * 16 + nidx) * 64 + 32 + lh * 16), c);
        vtanh(c);
        cstore(buf32, 32, nt * 16 + nidx, lh, c);
      }
    }
    LDS_FENCE();

    // (d) S_new = tanh(compressed @ Wex^T + b) -> allst[:,0:48]
    {
      const v16h ac = frag_a(buf32 + nidx * 32 + lh * 8);
#pragma unroll
      for (int nt = 0; nt < 3; ++nt) {
        v8f c = splat8(bias[B_E + nt * 16 + nidx]);
        c = wmma16(ac, frag_b(pWe + (nt * 16 + nidx) * 32 + lh * 16), c);
        vtanh(c);
        cstore(allst, 160, nt * 16 + nidx, lh, c);
      }
    }

    // (e,f,g) gates z,r over comb=[sensed|M]; h over combH=[sensed|r*M]; M update
    v8f zf[3], Mv[3], hf[3];
    {
      v16h am[3];
#pragma unroll
      for (int kc = 0; kc < 3; ++kc) am[kc] = frag_a(comb + nidx * 96 + kc * 32 + lh * 8);
#pragma unroll
      for (int nt = 0; nt < 3; ++nt) {
        v8f cz = splat8(bias[B_Z + nt * 16 + nidx]);
        v8f cr = splat8(bias[B_R + nt * 16 + nidx]);
#pragma unroll
        for (int kc = 0; kc < 3; ++kc) {
          cz = wmma16(am[kc], frag_b(pWz + (nt * 16 + nidx) * 96 + kc * 32 + lh * 16), cz);
          cr = wmma16(am[kc], frag_b(pWr + (nt * 16 + nidx) * 96 + kc * 32 + lh * 16), cr);
        }
        vsigm(cz); vsigm(cr);
        zf[nt] = cz;
        Mv[nt] = cload(comb, 96, 48 + nt * 16 + nidx, lh);
        v8f rm;
#pragma unroll
        for (int i = 0; i < 8; ++i) rm[i] = cr[i] * Mv[nt][i];
        cstore(combH, 96, 48 + nt * 16 + nidx, lh, rm);
      }
      LDS_FENCE();
      v16h ah[3];
#pragma unroll
      for (int kc = 0; kc < 3; ++kc) ah[kc] = frag_a(combH + nidx * 96 + kc * 32 + lh * 8);
#pragma unroll
      for (int nt = 0; nt < 3; ++nt) {
        v8f ch = splat8(bias[B_H + nt * 16 + nidx]);
#pragma unroll
        for (int kc = 0; kc < 3; ++kc)
          ch = wmma16(ah[kc], frag_b(pWh + (nt * 16 + nidx) * 96 + kc * 32 + lh * 16), ch);
        vtanh(ch);
        hf[nt] = ch;
      }
#pragma unroll
      for (int nt = 0; nt < 3; ++nt) {
        v8f mn;
#pragma unroll
        for (int i = 0; i < 8; ++i)
          mn[i] = (1.0f - zf[nt][i]) * Mv[nt][i] + zf[nt][i] * hf[nt][i];
        cstore(comb,  96,  48 + nt * 16 + nidx, lh, mn);  // M for next step
        cstore(allst, 160, 48 + nt * 16 + nidx, lh, mn);  // M_new in allst
      }
    }
    LDS_FENCE();

    // (h) D_new = tanh(allst@Wi^T+b) * sigmoid(allst@Wp^T+b); allst has OLD Dst
    {
      v16h aa[5];
#pragma unroll
      for (int kc = 0; kc < 5; ++kc) aa[kc] = frag_a(allst + nidx * 160 + kc * 32 + lh * 8);
#pragma unroll
      for (int nt = 0; nt < 3; ++nt) {
        v8f ci = splat8(bias[B_I + nt * 16 + nidx]);
        v8f cp = splat8(bias[B_P + nt * 16 + nidx]);
#pragma unroll
        for (int kc = 0; kc < 5; ++kc) {
          ci = wmma16(aa[kc], frag_b(pWi + (nt * 16 + nidx) * 160 + kc * 32 + lh * 16), ci);
          cp = wmma16(aa[kc], frag_b(pWp + (nt * 16 + nidx) * 160 + kc * 32 + lh * 16), cp);
        }
        v8f dn;
#pragma unroll
        for (int i = 0; i < 8; ++i) dn[i] = tanh_f(ci[i]) * sigm_f(cp[i]);
        cstore(allst, 160, 96 + nt * 16 + nidx, lh, dn);   // becomes Dst next step
      }
    }
    LDS_FENCE();

    // (i) action = D_new @ Wo^T + b  (N padded 4->16, K 48->64)
    {
      const v16h ad0 = frag_a(allst + nidx * 160 + 96  + lh * 8);
      const v16h ad1 = frag_a(allst + nidx * 160 + 128 + lh * 8);
      v8f co = splat8(bias[B_O + nidx]);
      co = wmma16(ad0, frag_b(pWo + nidx * 64 + 0  + lh * 16), co);
      co = wmma16(ad1, frag_b(pWo + nidx * 64 + 32 + lh * 16), co);
      // Unconditional LDS use keeps WMMA out of the divergent region (EXEC all-1s).
      cstore(buf32, 32, nidx, lh, co);
      if (nidx < ODIM) {
#pragma unroll
        for (int i = 0; i < 8; ++i)
          out[(((size_t)(row0 + lh * 8 + i)) * SEQ + (size_t)t) * ODIM + nidx] = co[i];
      }
    }
  }
}

extern "C" void kernel_launch(void* const* d_in, const int* in_sizes, int n_in,
                              void* d_out, int out_size, void* d_ws, size_t ws_size,
                              hipStream_t stream) {
  (void)in_sizes; (void)n_in; (void)out_size; (void)d_ws; (void)ws_size;
  hipFuncSetAttribute(reinterpret_cast<const void*>(&anima_scan_kernel),
                      hipFuncAttributeMaxDynamicSharedMemorySize, LDS_BYTES);
  anima_scan_kernel<<<dim3(NBLOCKS), dim3(TPB), LDS_BYTES, stream>>>(
      (const float*)d_in[0],
      (const float*)d_in[1],  (const float*)d_in[2],
      (const float*)d_in[3],  (const float*)d_in[4],
      (const float*)d_in[5],  (const float*)d_in[6],
      (const float*)d_in[7],  (const float*)d_in[8],
      (const float*)d_in[9],  (const float*)d_in[10],
      (const float*)d_in[11], (const float*)d_in[12],
      (const float*)d_in[13], (const float*)d_in[14],
      (const float*)d_in[15], (const float*)d_in[16],
      (const float*)d_in[17], (const float*)d_in[18],
      (float*)d_out);
}
